// MultiHeadAttention_82411832476301
// MI455X (gfx1250) — matrix operand order repaired and checked by reading, not compiled, not run on hardware
//
#include <hip/hip_runtime.h>
#include <cstdint>

// ---------------------------------------------------------------------------
// MHA forward for MI455X (gfx1250): f16 WMMA everywhere, f32 accumulation.
// GEMM tiles staged by the Tensor Data Mover (tensor_load_to_lds);
// flash-attention tiles staged with global_load_async_to_lds_b128.
// BS=4, QLEN=2048, DIM=1024, HEADS=16, HD=64.
// ---------------------------------------------------------------------------

#define QLEN  2048
#define DIMSZ 1024
#define HEADS 16
#define HD    64
#define MTOT  8192   // BS*QLEN

typedef _Float16 half4  __attribute__((ext_vector_type(4)));
typedef _Float16 half8  __attribute__((ext_vector_type(8)));
typedef _Float16 half16 __attribute__((ext_vector_type(16)));
typedef float    float8 __attribute__((ext_vector_type(8)));
typedef unsigned int u32x4 __attribute__((ext_vector_type(4)));
typedef int      i32x8 __attribute__((ext_vector_type(8)));
typedef int      i32x4 __attribute__((ext_vector_type(4)));

__device__ __forceinline__ half16 cat8(half8 a, half8 b) {
  return __builtin_shufflevector(a, b, 0,1,2,3,4,5,6,7,8,9,10,11,12,13,14,15);
}
// A-matrix 16xK fragment: lane row = M, halves[0..7]=K@(lhi*8), halves[8..15]=K@(lhi*8+16)
__device__ __forceinline__ half16 fragA(const _Float16* p) {
  return cat8(*(const half8*)p, *(const half8*)(p + 16));
}
// B-matrix Kx16 fragment from [N][K] LDS layout: lane row = N, halves[0..7]=K@(lhi*8), halves[8..15]=K@(lhi*8+16)
__device__ __forceinline__ half16 fragB(const _Float16* p) {
  return cat8(*(const half8*)p, *(const half8*)(p + 16));
}
__device__ __forceinline__ float8 wmma32(half16 a, half16 b, float8 c) {
  return __builtin_amdgcn_wmma_f32_16x16x32_f16(false, a, false, b, (short)0, c,
                                                false, false);
}
// CDNA5 async global->LDS copy (bypasses VGPRs), tracked by ASYNCcnt.
__device__ __forceinline__ void async_b128(uint32_t lds, const _Float16* g) {
  asm volatile("global_load_async_to_lds_b128 %0, %1, off"
               :: "v"(lds), "v"(g) : "memory");
}
__device__ __forceinline__ void wait_async0() {
  asm volatile("s_wait_asynccnt 0" ::: "memory");
}
__device__ __forceinline__ void compiler_fence() {
  asm volatile("" ::: "memory");
}

// Fully-unrolled async tile stage: ROWS x ROWHALVES f16, 16B chunks, no
// divergent control flow (ROWS*ROWHALVES/8 must be a multiple of 256).
template <int ROWS, int ROWHALVES>
__device__ __forceinline__ void stage_async_t(const _Float16* __restrict__ g,
                                              int gstride, _Float16* l, int lstride) {
  constexpr int CPR = ROWHALVES / 8;             // 16B chunks per row
  constexpr int ITERS = (ROWS * CPR) / 256;
#pragma unroll
  for (int i = 0; i < ITERS; ++i) {
    const int c = threadIdx.x + i * 256;
    const int r = c / CPR;
    const int k = (c % CPR) * 8;
    async_b128((uint32_t)(uintptr_t)(l + r * lstride + k),
               g + (size_t)r * gstride + k);
  }
}

// Tensor Data Mover 2D descriptor (tileW x tileH elements of 2 bytes, row
// stride strideElems) per CDNA5 ISA ch.8 (count=1, type=2). The descriptor is
// built in uniform code so it lands in SGPRs; only the issue is wave-gated.
struct TdmDesc { u32x4 g0; i32x8 g1; };

__device__ __forceinline__ TdmDesc tdm_make_2d(uint32_t lds, const void* gptr,
                                               uint32_t tileW, uint32_t tileH,
                                               uint32_t strideElems) {
  const uint64_t ga = (uint64_t)(uintptr_t)gptr;
  TdmDesc d;
  d.g0[0] = 1u;                                           // count=1, user mode
  d.g0[1] = lds;                                          // lds_addr
  d.g0[2] = (uint32_t)ga;                                 // global_addr[31:0]
  d.g0[3] = (uint32_t)((ga >> 32) & 0x01FFFFFFu) | (2u << 30);  // ga[56:32], type=2
  d.g1[0] = (int)(1u << 16);                              // data_size=1 (2 bytes)
  d.g1[1] = (int)((tileW & 0xFFFFu) << 16);               // tensor_dim0 lo16
  d.g1[2] = (int)((tileW >> 16) | ((tileH & 0xFFFFu) << 16)); // t_dim0 hi / t_dim1 lo
  d.g1[3] = (int)((tileH >> 16) | ((tileW & 0xFFFFu) << 16)); // t_dim1 hi / tile_dim0
  d.g1[4] = (int)(tileH & 0xFFFFu);                       // tile_dim1 ; tile_dim2=0
  d.g1[5] = (int)strideElems;                             // tensor_dim0_stride lo32
  d.g1[6] = 0;                                            // stride0 hi / stride1 lo
  d.g1[7] = 0;                                            // stride1 hi
  return d;
}
__device__ __forceinline__ void tdm_issue(const TdmDesc& d) {
  const i32x4 z4 = {0, 0, 0, 0};
  const i32x8 z8 = {0, 0, 0, 0, 0, 0, 0, 0};
  __builtin_amdgcn_tensor_load_to_lds(d.g0, d.g1, z4, z4, z8, 0);
}

// ---------------------------------------------------------------------------
__global__ __launch_bounds__(256) void cast_f32_to_f16(const float* __restrict__ s,
                                                       _Float16* __restrict__ d,
                                                       int n) {
  int i = (blockIdx.x * 256 + threadIdx.x) * 4;
  if (i < n) {
    float4 v = *(const float4*)(s + i);
    half4 h = { (_Float16)v.x, (_Float16)v.y, (_Float16)v.z, (_Float16)v.w };
    *(half4*)(d + i) = h;
  }
}

// ---------------------------------------------------------------------------
// C = A[M,1024] * W[1024(e),1024(d)]^T + bias, scaled. Tiles staged by TDM.
// out16 != null : scatter f16 into [b,h,s,hd] QKV layout.
// out32 != null : plain f32 [M,1024] (final O projection).
__global__ __launch_bounds__(256) void gemm_wmma(
    const _Float16* __restrict__ A,
    const _Float16* __restrict__ W,
    const float* __restrict__ bias,
    _Float16* __restrict__ out16,
    float* __restrict__ out32,
    float scale) {
  constexpr int BM = 128, BN = 128, BK = 32, NK = DIMSZ / BK;
  __shared__ __align__(16) _Float16 aT[2][BM][BK];
  __shared__ __align__(16) _Float16 bT[2][BN][BK];

  const int m0 = blockIdx.y * BM;
  const int n0 = blockIdx.x * BN;
  const int lane = threadIdx.x & 31;
  const int wid  = threadIdx.x >> 5;
  const int wm   = (wid & 3) * 32;   // 4 waves tile M
  const int wn   = (wid >> 2) * 64;  // 2 waves tile N
  const int lrow = lane & 15;
  const int lhi  = lane >> 4;
  const bool tdmw = (wid == 0);      // wave 0 drives the Tensor Data Mover

  const float8 f8z = {0.f,0.f,0.f,0.f,0.f,0.f,0.f,0.f};
  float8 acc[2][4];
#pragma unroll
  for (int mt = 0; mt < 2; ++mt)
#pragma unroll
    for (int nt = 0; nt < 4; ++nt) acc[mt][nt] = f8z;

  const _Float16* Ab = A + (size_t)m0 * DIMSZ;
  const _Float16* Wb = W + (size_t)n0 * DIMSZ;
  const uint32_t ldsA[2] = { (uint32_t)(uintptr_t)&aT[0][0][0],
                             (uint32_t)(uintptr_t)&aT[1][0][0] };
  const uint32_t ldsB[2] = { (uint32_t)(uintptr_t)&bT[0][0][0],
                             (uint32_t)(uintptr_t)&bT[1][0][0] };

  {
    const TdmDesc dA = tdm_make_2d(ldsA[0], Ab, BK, BM, DIMSZ);
    const TdmDesc dB = tdm_make_2d(ldsB[0], Wb, BK, BN, DIMSZ);
    if (tdmw) { tdm_issue(dA); tdm_issue(dB); }
  }

  int buf = 0;
  for (int kt = 0; kt < NK; ++kt) {
    // No-op for the 7 non-TDM waves (their TENSORcnt is always 0), so no
    // divergent wrapper is needed.
    __builtin_amdgcn_s_wait_tensorcnt(0);
    __syncthreads();
    {
      const int k0 = (kt + 1) * BK;  // address math only if last iteration
      const TdmDesc dA = tdm_make_2d(ldsA[buf ^ 1], Ab + k0, BK, BM, DIMSZ);
      const TdmDesc dB = tdm_make_2d(ldsB[buf ^ 1], Wb + k0, BK, BN, DIMSZ);
      if ((kt + 1 < NK) && tdmw) { tdm_issue(dA); tdm_issue(dB); }
    }
    half16 af[2], bf[4];
#pragma unroll
    for (int mt = 0; mt < 2; ++mt)
      af[mt] = fragA(&aT[buf][wm + mt * 16 + lrow][lhi * 8]);
#pragma unroll
    for (int nt = 0; nt < 4; ++nt)
      bf[nt] = fragB(&bT[buf][wn + nt * 16 + lrow][lhi * 8]);
#pragma unroll
    for (int mt = 0; mt < 2; ++mt)
#pragma unroll
      for (int nt = 0; nt < 4; ++nt)
        acc[mt][nt] = wmma32(af[mt], bf[nt], acc[mt][nt]);
    buf ^= 1;
  }

#pragma unroll
  for (int mt = 0; mt < 2; ++mt) {
#pragma unroll
    for (int nt = 0; nt < 4; ++nt) {
      const int col = n0 + wn + nt * 16 + lrow;
      const float bv = bias[col];
#pragma unroll
      for (int v = 0; v < 8; ++v) {
        const int row = m0 + wm + mt * 16 + v + 8 * lhi;
        const float val = (acc[mt][nt][v] + bv) * scale;
        if (out32) {
          out32[(size_t)row * DIMSZ + col] = val;
        } else {
          const int bb = row >> 11, s = row & (QLEN - 1);
          const int h = col >> 6, hd = col & (HD - 1);
          out16[(((size_t)bb * HEADS + h) * QLEN + s) * HD + hd] = (_Float16)val;
        }
      }
    }
  }
}

// ---------------------------------------------------------------------------
// Flash attention over f16 Q,K,V in [b,h,s,hd]; f32 online softmax; f16 ctx out
// in [b*qlen, dim] with column h*64+hd. K/Q staged via async-to-LDS.
__global__ __launch_bounds__(256) void flash_attn(
    const _Float16* __restrict__ Q,
    const _Float16* __restrict__ K,
    const _Float16* __restrict__ V,
    const int* __restrict__ mask,
    _Float16* __restrict__ ctx) {
  constexpr int BQ = 128, BJ = 64, NJ = QLEN / BJ;
  __shared__ __align__(16) union {
    _Float16 q[BQ][HD];        // Q tile (only needed until aQ frags are loaded)
    _Float16 p[8][16][72];     // per-wave P staging (C-layout -> A-layout)
  } qp;
  __shared__ __align__(16) _Float16 kT[2][BJ][HD];  // [j][hd]  ([N][K] for QK^T)
  __shared__ __align__(16) _Float16 vT[2][HD][BJ];  // [hd][j]  ([N][K] for P*V)
  __shared__ float mbias[QLEN];                     // additive mask bias table

  const int bh = blockIdx.y;
  const int b = bh >> 4, h = bh & 15;
  const int q0 = blockIdx.x * BQ;
  const size_t head_off = (size_t)bh * QLEN * HD;
  const _Float16* Qg = Q + head_off + (size_t)q0 * HD;
  const _Float16* Kg = K + head_off;
  const _Float16* Vg = V + head_off;

  const int lane = threadIdx.x & 31;
  const int wid  = threadIdx.x >> 5;
  const int lrow = lane & 15;
  const int lhi  = lane >> 4;

  const float8 f8z = {0.f,0.f,0.f,0.f,0.f,0.f,0.f,0.f};
  float8 o[4];
#pragma unroll
  for (int nt = 0; nt < 4; ++nt) o[nt] = f8z;
  float m_i[8], l_i[8];
#pragma unroll
  for (int v = 0; v < 8; ++v) { m_i[v] = -3.0e38f; l_i[v] = 0.f; }

  // V tile staged transposed: [j][hd] global -> [hd][j] LDS.
  auto stage_v = [&](const _Float16* vg, _Float16 (*vt)[BJ]) {
    const int t = threadIdx.x;
    const int j = t >> 2;
    const int hd0 = (t & 3) * 16;
    const _Float16* p = vg + (size_t)j * HD + hd0;
    half8 u0 = *(const half8*)p;
    half8 u1 = *(const half8*)(p + 8);
#pragma unroll
    for (int i = 0; i < 8; ++i) {
      vt[hd0 + i][j]     = u0[i];
      vt[hd0 + 8 + i][j] = u1[i];
    }
  };

  // Prologue: Q tile + first K/V tile + mask bias table.
  stage_async_t<BQ, HD>(Qg, HD, &qp.q[0][0], HD);
  stage_async_t<BJ, HD>(Kg, HD, &kT[0][0][0], HD);
  stage_v(Vg, vT[0]);
#pragma unroll
  for (int i = 0; i < QLEN / 256; ++i) {
    const int j = threadIdx.x + i * 256;
    mbias[j] = (mask[b * QLEN + j] == 0) ? -1.0e30f : 0.0f;
  }
  wait_async0();
  __syncthreads();

  half16 aQ[2];
#pragma unroll
  for (int kk = 0; kk < 2; ++kk)
    aQ[kk] = fragA(&qp.q[wid * 16 + lrow][kk * 32 + lhi * 8]);
  __syncthreads();  // qp.q space is reused as qp.p below

  int buf = 0;
  for (int jt = 0; jt < NJ; ++jt) {
    if (jt + 1 < NJ) {
      stage_async_t<BJ, HD>(Kg + (size_t)(jt + 1) * BJ * HD, HD, &kT[buf ^ 1][0][0], HD);
      stage_v(Vg + (size_t)(jt + 1) * BJ * HD, vT[buf ^ 1]);
    }
    // S = Q * K^T  (16x64 per wave, K-dim = hd = 64)
    float8 sc[4];
#pragma unroll
    for (int nt = 0; nt < 4; ++nt) {
      sc[nt] = f8z;
#pragma unroll
      for (int kk = 0; kk < 2; ++kk) {
        half16 bf = fragB(&kT[buf][nt * 16 + lrow][kk * 32 + lhi * 8]);
        sc[nt] = wmma32(aQ[kk], bf, sc[nt]);
      }
    }
    // padding mask (reference: mask==0 -> -inf), from the LDS bias table
#pragma unroll
    for (int nt = 0; nt < 4; ++nt) {
      const float mb = mbias[jt * BJ + nt * 16 + lrow];
#pragma unroll
      for (int v = 0; v < 8; ++v) sc[nt][v] += mb;
    }
    // online softmax; row r = v + 8*lhi, cols spread over 16 lanes x 4 tiles
#pragma unroll
    for (int v = 0; v < 8; ++v) {
      float mx = fmaxf(fmaxf(sc[0][v], sc[1][v]), fmaxf(sc[2][v], sc[3][v]));
#pragma unroll
      for (int off = 1; off < 16; off <<= 1)
        mx = fmaxf(mx, __shfl_xor(mx, off, 16));
      const float mnew = fmaxf(m_i[v], mx);
      const float corr = __expf(m_i[v] - mnew);
      m_i[v] = mnew;
      float rs = 0.f;
#pragma unroll
      for (int nt = 0; nt < 4; ++nt) {
        const float p = __expf(sc[nt][v] - mnew);
        sc[nt][v] = p;
        rs += p;
      }
#pragma unroll
      for (int off = 1; off < 16; off <<= 1)
        rs += __shfl_xor(rs, off, 16);
      l_i[v] = l_i[v] * corr + rs;
#pragma unroll
      for (int nt = 0; nt < 4; ++nt) o[nt][v] *= corr;
    }
    // P (C-layout regs) -> per-wave LDS -> A-layout frags
#pragma unroll
    for (int nt = 0; nt < 4; ++nt)
#pragma unroll
      for (int v = 0; v < 8; ++v)
        qp.p[wid][v + 8 * lhi][nt * 16 + lrow] = (_Float16)sc[nt][v];
    compiler_fence();
    // O += P * V  (K-dim = j = 64)
#pragma unroll
    for (int kk = 0; kk < 2; ++kk) {
      half16 aP = fragA(&qp.p[wid][lrow][kk * 32 + lhi * 8]);
#pragma unroll
      for (int nt = 0; nt < 4; ++nt) {
        half16 bf = fragB(&vT[buf][nt * 16 + lrow][kk * 32 + lhi * 8]);
        o[nt] = wmma32(aP, bf, o[nt]);
      }
    }
    wait_async0();
    __syncthreads();
    buf ^= 1;
  }

  // normalize and store context
#pragma unroll
  for (int v = 0; v < 8; ++v) {
    const float inv = (l_i[v] > 0.f) ? 1.0f / l_i[v] : 0.f;
    const int s = q0 + wid * 16 + v + 8 * lhi;
    const size_t rowbase = ((size_t)b * QLEN + s) * DIMSZ + h * HD;
#pragma unroll
    for (int nt = 0; nt < 4; ++nt)
      ctx[rowbase + nt * 16 + lrow] = (_Float16)(o[nt][v] * inv);
  }
}

// ---------------------------------------------------------------------------
extern "C" void kernel_launch(void* const* d_in, const int* in_sizes, int n_in,
                              void* d_out, int out_size, void* d_ws, size_t ws_size,
                              hipStream_t stream) {
  (void)in_sizes; (void)n_in; (void)out_size; (void)ws_size;
  const float* input = (const float*)d_in[0];
  const int*   mask  = (const int*)d_in[1];
  const float* wq = (const float*)d_in[2];
  const float* bq = (const float*)d_in[3];
  const float* wk = (const float*)d_in[4];
  const float* bk = (const float*)d_in[5];
  const float* wv = (const float*)d_in[6];
  const float* bv = (const float*)d_in[7];
  const float* wo = (const float*)d_in[8];
  const float* bo = (const float*)d_in[9];
  float* out = (float*)d_out;

  char* ws = (char*)d_ws;
  const size_t MB = 1024 * 1024;
  _Float16* Xh  = (_Float16*)(ws);             // 16 MB
  _Float16* Wqh = (_Float16*)(ws + 16 * MB);   //  2 MB each
  _Float16* Wkh = (_Float16*)(ws + 18 * MB);
  _Float16* Wvh = (_Float16*)(ws + 20 * MB);
  _Float16* Woh = (_Float16*)(ws + 22 * MB);
  _Float16* Qh  = (_Float16*)(ws + 24 * MB);   // 16 MB each, [b,h,s,hd]
  _Float16* Kh  = (_Float16*)(ws + 40 * MB);
  _Float16* Vh  = (_Float16*)(ws + 56 * MB);
  _Float16* Ch  = (_Float16*)(ws + 72 * MB);   // 16 MB, [m, dim]

  const int nX = MTOT * DIMSZ;
  const int nW = DIMSZ * DIMSZ;
  cast_f32_to_f16<<<nX / 1024, 256, 0, stream>>>(input, Xh, nX);
  cast_f32_to_f16<<<nW / 1024, 256, 0, stream>>>(wq, Wqh, nW);
  cast_f32_to_f16<<<nW / 1024, 256, 0, stream>>>(wk, Wkh, nW);
  cast_f32_to_f16<<<nW / 1024, 256, 0, stream>>>(wv, Wvh, nW);
  cast_f32_to_f16<<<nW / 1024, 256, 0, stream>>>(wo, Woh, nW);

  dim3 ggrid(DIMSZ / 128, MTOT / 128);          // (8, 64)
  gemm_wmma<<<ggrid, 256, 0, stream>>>(Xh, Wqh, bq, Qh, nullptr, 0.125f); // 1/sqrt(64)
  gemm_wmma<<<ggrid, 256, 0, stream>>>(Xh, Wkh, bk, Kh, nullptr, 1.0f);
  gemm_wmma<<<ggrid, 256, 0, stream>>>(Xh, Wvh, bv, Vh, nullptr, 1.0f);

  dim3 fgrid(QLEN / 128, 4 * HEADS);            // (16, 64)
  flash_attn<<<fgrid, 256, 0, stream>>>(Qh, Kh, Vh, mask, Ch);

  gemm_wmma<<<ggrid, 256, 0, stream>>>(Ch, Woh, bo, nullptr, out, 1.0f);
}
